// Custom_Linear_13228499272227
// MI455X (gfx1250) — 8-bit matrix operand order repaired and checked by reading, compiled, not run on hardware
//
#include <hip/hip_runtime.h>
#include <stdint.h>

typedef __attribute__((ext_vector_type(8))) int v8i;
typedef __attribute__((ext_vector_type(4))) int v4i;
typedef __attribute__((ext_vector_type(2))) int v2i;

#define TDIM 2048
#define KDIM 4096
#define NDIM 11008

#define BM 128
#define BN 128
#define BK 64
#define LDSB 80
#define NBUF 3
#define BUFS (BM * LDSB)
#define THREADS 256
#define KT (KDIM / BK)

typedef __attribute__((address_space(3))) char lds_char;

template<int I> struct ic { static constexpr int v = I; };

__device__ __forceinline__ uint32_t lds_off(const void* p) {
    return (uint32_t)(size_t)(lds_char*)(char*)p;
}

__device__ __forceinline__ void async_ld16(uint32_t lds_byte_off, uint32_t voff,
                                           const void* sbase) {
    asm volatile("global_load_async_to_lds_b128 %0, %1, %2"
                 :: "v"(lds_byte_off), "v"(voff), "s"(sbase)
                 : "memory");
}

__device__ __forceinline__ void wait_async0() {
    asm volatile("s_wait_asynccnt 0x0" ::: "memory");
}
__device__ __forceinline__ void wait_async4() {
    asm volatile("s_wait_asynccnt 0x4" ::: "memory");
}

__global__ void __launch_bounds__(THREADS)
i8gemm_requant(const int8_t* __restrict__ x,
               const int8_t* __restrict__ w,
               const float*  __restrict__ scale_x,
               const float*  __restrict__ scale_w,
               const float*  __restrict__ scale_y,
               float*        __restrict__ out)
{
    __shared__ __align__(16) char sA[NBUF][BUFS];
    __shared__ __align__(16) char sB[NBUF][BUFS];
    __shared__ float sRS[BM];

    const int tid   = threadIdx.x;
    const int lane  = tid & 31;
    const int wave  = tid >> 5;
    const int waveM = wave >> 1;
    const int waveN = wave & 1;
    const int lhalf = lane >> 4;
    const int l15   = lane & 15;

    const int blockM = blockIdx.y * BM;
    const int blockN = blockIdx.x * BN;

    if (tid < BM) {
        int r = blockM + tid;
        sRS[tid] = scale_x[r] * scale_w[0] / scale_y[r];
    }

    const int c0  = tid & 3;
    const int rA0 = tid >> 2;
    const int rA1 = 64 + (tid >> 2);
    const int8_t* gA = x + (size_t)blockM * KDIM;
    const int8_t* gB = w + (size_t)blockN * KDIM;

    const uint32_t va0 = (uint32_t)(rA0 * KDIM + c0 * 16);
    const uint32_t va1 = (uint32_t)(rA1 * KDIM + c0 * 16);

    uint32_t lA0[NBUF], lA1[NBUF], lB0[NBUF], lB1[NBUF];
    {
        const uint32_t a0 = lds_off(&sA[0][rA0 * LDSB + c0 * 16]);
        const uint32_t a1 = lds_off(&sA[0][rA1 * LDSB + c0 * 16]);
        const uint32_t b0 = lds_off(&sB[0][rA0 * LDSB + c0 * 16]);
        const uint32_t b1 = lds_off(&sB[0][rA1 * LDSB + c0 * 16]);
#pragma unroll
        for (int b = 0; b < NBUF; ++b) {
            lA0[b] = a0 + b * BUFS;  lA1[b] = a1 + b * BUFS;
            lB0[b] = b0 + b * BUFS;  lB1[b] = b1 + b * BUFS;
        }
    }

    const char* baseA_t = &sA[0][(waveM * 32 + l15) * LDSB + lhalf * 8];
    const char* baseB_t = &sB[0][(waveN * 64 + l15) * LDSB + lhalf * 8];

    v8i acc[2][4];
#pragma unroll
    for (int m = 0; m < 2; ++m)
#pragma unroll
        for (int n = 0; n < 4; ++n)
            acc[m][n] = (v8i)(0);

    auto compute_tile = [&](auto BUFc) {
        constexpr int BO = decltype(BUFc)::v * BUFS;
        v8i af[2];
#pragma unroll
        for (int m = 0; m < 2; ++m) {
            const char* ap = baseA_t + BO + m * (16 * LDSB);
            v2i q0 = *(const v2i*)(ap +  0);
            v2i q1 = *(const v2i*)(ap + 16);
            v2i q2 = *(const v2i*)(ap + 32);
            v2i q3 = *(const v2i*)(ap + 48);
            v8i a = { q0.x, q0.y, q1.x, q1.y, q2.x, q2.y, q3.x, q3.y };
            af[m] = a;
        }
#pragma unroll
        for (int n = 0; n < 4; ++n) {
            const char* bp = baseB_t + BO + n * (16 * LDSB);
            v2i r0 = *(const v2i*)(bp +  0);
            v2i r1 = *(const v2i*)(bp + 16);
            v2i r2 = *(const v2i*)(bp + 32);
            v2i r3 = *(const v2i*)(bp + 48);
            v8i bf = { r0.x, r0.y, r1.x, r1.y, r2.x, r2.y, r3.x, r3.y };
#pragma unroll
            for (int m = 0; m < 2; ++m)
                acc[m][n] = __builtin_amdgcn_wmma_i32_16x16x64_iu8(
                    true, af[m], true, bf, acc[m][n], false, false);
        }
    };

    async_ld16(lA0[0], va0, gA);
    async_ld16(lA1[0], va1, gA);
    async_ld16(lB0[0], va0, gB);
    async_ld16(lB1[0], va1, gB);
    async_ld16(lA0[1], va0, gA + BK);
    async_ld16(lA1[1], va1, gA + BK);
    async_ld16(lB0[1], va0, gB + BK);
    async_ld16(lB1[1], va1, gB + BK);

    const int8_t* pA = gA + 2 * BK;
    const int8_t* pB = gB + 2 * BK;

    auto pipe_step = [&](auto CURc, auto NXTc) {
        constexpr int NXT = decltype(NXTc)::v;
        wait_async4();
        __syncthreads();
        async_ld16(lA0[NXT], va0, pA);
        async_ld16(lA1[NXT], va1, pA);
        async_ld16(lB0[NXT], va0, pB);
        async_ld16(lB1[NXT], va1, pB);
        pA += BK; pB += BK;
        compute_tile(CURc);
    };

    static_assert(KT == 64, "pipeline schedule assumes KT == 64");
#pragma unroll 1
    for (int g = 0; g < (KT - 4) / 3; ++g) {
        pipe_step(ic<0>{}, ic<2>{});
        pipe_step(ic<1>{}, ic<0>{});
        pipe_step(ic<2>{}, ic<1>{});
    }
    pipe_step(ic<0>{}, ic<2>{});
    pipe_step(ic<1>{}, ic<0>{});
    wait_async4();
    __syncthreads();
    compute_tile(ic<2>{});
    wait_async0();
    __syncthreads();
    compute_tile(ic<0>{});

    const uint32_t colB = (uint32_t)(blockN + waveN * 64 + l15);
#pragma unroll
    for (int m = 0; m < 2; ++m) {
#pragma unroll
        for (int n = 0; n < 4; ++n) {
#pragma unroll
            for (int i = 0; i < 8; ++i) {
                const int rLoc = waveM * 32 + m * 16 + lhalf * 8 + i;
                float v = (float)acc[m][n][i] * sRS[rLoc];
                v = __builtin_rintf(v);
                v = fminf(127.0f, fmaxf(-128.0f, v));
                const uint32_t idx = (uint32_t)(blockM + rLoc) * (uint32_t)NDIM
                                   + colB + (uint32_t)(n * 16);
                out[idx] = v;
            }
        }
    }
}

__global__ void copy_scale_y(const float* __restrict__ sy,
                             float* __restrict__ dst, int T)
{
    int i = blockIdx.x * blockDim.x + threadIdx.x;
    if (i < T) dst[i] = sy[i];
}

extern "C" void kernel_launch(void* const* d_in, const int* in_sizes, int n_in,
                              void* d_out, int out_size, void* d_ws, size_t ws_size,
                              hipStream_t stream)
{
    (void)n_in; (void)in_sizes; (void)d_ws; (void)ws_size; (void)out_size;

    const int8_t* x  = (const int8_t*)d_in[0];
    const int8_t* w  = (const int8_t*)d_in[1];
    const float*  sx = (const float*)d_in[2];
    const float*  sw = (const float*)d_in[3];
    const float*  sy = (const float*)d_in[4];
    float* out = (float*)d_out;

    dim3 grid(NDIM / BN, TDIM / BM);
    i8gemm_requant<<<grid, THREADS, 0, stream>>>(x, w, sx, sw, sy, out);

    copy_scale_y<<<(TDIM + 255) / 256, 256, 0, stream>>>(sy, out + (size_t)TDIM * NDIM, TDIM);
}
